// DGCNN_9801115369861
// MI455X (gfx1250) — matrix operand order repaired and checked by reading, not compiled, not run on hardware
//
#include <hip/hip_runtime.h>
#include <hip/hip_bf16.h>

#define BATCH 8
#define NPTS  2048
#define KNN   20
#define SLOPE 0.2f
#define CCAT  512
#define FDIM  1024
#define WPB   4          // waves per block in final GEMM
#define RT    (NPTS/16)  // 128 row tiles

typedef float  v2f   __attribute__((ext_vector_type(2)));
typedef float  v8f   __attribute__((ext_vector_type(8)));
typedef __bf16 v8bf  __attribute__((ext_vector_type(8)));
typedef __bf16 v16bf __attribute__((ext_vector_type(16)));

__device__ __forceinline__ float leaky(float z) { return z > 0.f ? z : SLOPE * z; }

__device__ __forceinline__ v16bf frag16(v8bf lo, v8bf hi) {
  return __builtin_shufflevector(lo, hi, 0, 1, 2, 3, 4, 5, 6, 7,
                                 8, 9, 10, 11, 12, 13, 14, 15);
}

// ---------------- squared norms per point ----------------
template<int C>
__global__ void norms_kernel(const float* __restrict__ X, float* __restrict__ xx) {
  int p = blockIdx.x * blockDim.x + threadIdx.x;   // 0..B*N-1
  const float* r = X + (size_t)p * C;
  float s = 0.f;
  #pragma unroll
  for (int c = 0; c < C; ++c) { float v = r[c]; s += v * v; }
  xx[p] = s;
}

// ---------------- weights -> zero-padded bf16 ----------------
__global__ void convw_kernel(const float* __restrict__ W, __bf16* __restrict__ Wb,
                             int C2, int COLS, int total) {
  int i = blockIdx.x * blockDim.x + threadIdx.x;
  if (i >= total) return;
  int o = i / COLS, c = i % COLS;
  Wb[i] = (c < C2) ? (__bf16)W[(size_t)o * C2 + c] : (__bf16)0.f;
}

// ---------------- 16x16 distance tile via fp32 WMMA ----------------
// One wave per tile. d2[i][j] = |xi|^2 - 2 xi.xj + |xj|^2
template<int C>
__global__ void dist_kernel(const float* __restrict__ Xb, const float* __restrict__ xxb,
                            float* __restrict__ d2) {
  int c0 = blockIdx.x * 16, r0 = blockIdx.y * 16;
  int lane = threadIdx.x & 31;
  int m  = lane & 15;
  int kb = (lane & 16) ? 2 : 0;       // f32 A/B layout: lanes 16-31 hold K pair {2,3}
  v8f acc = {0.f,0.f,0.f,0.f,0.f,0.f,0.f,0.f};
  const float* Arow = Xb + (size_t)(r0 + m) * C;  // A: row = lane%16
  const float* Brow = Xb + (size_t)(c0 + m) * C;  // B: col = lane%16
  if constexpr ((C & 3) == 0) {
    #pragma unroll
    for (int kk = 0; kk < C; kk += 4) {
      v2f a = *(const v2f*)(Arow + kk + kb);      // global_load_b64, no guards
      v2f b = *(const v2f*)(Brow + kk + kb);
      acc = __builtin_amdgcn_wmma_f32_16x16x4_f32(false, a, false, b, (short)0, acc,
                                                  false, false);
    }
  } else {                                        // C == 3: single chunk, branchless pad
    bool ok = (kb + 1) < C;                       // true only for lanes 0-15
    v2f a, b;
    a.x = Arow[kb];
    b.x = Brow[kb];
    float ay = Arow[ok ? kb + 1 : 0];
    float by = Brow[ok ? kb + 1 : 0];
    a.y = ok ? ay : 0.f;
    b.y = ok ? by : 0.f;
    acc = __builtin_amdgcn_wmma_f32_16x16x4_f32(false, a, false, b, (short)0, acc,
                                                false, false);
  }
  float xc = xxb[c0 + m];
  int moff = (lane & 16) ? 8 : 0;     // C/D layout: VGPR e -> M = e (+8 for hi lanes)
  #pragma unroll
  for (int e = 0; e < 8; ++e) {
    int row = r0 + e + moff;
    d2[(size_t)row * NPTS + (c0 + m)] = xxb[row] - 2.f * acc[e] + xc;
  }
}

// ---------------- top-K (K=20) smallest per row ----------------
__global__ void topk_kernel(const float* __restrict__ d2, int* __restrict__ idxb) {
  int i = blockIdx.x * blockDim.x + threadIdx.x;  // row 0..N-1
  const float* row = d2 + (size_t)i * NPTS;
  float bd[KNN]; int bi[KNN];
  #pragma unroll
  for (int t = 0; t < KNN; ++t) { bd[t] = 3.4e38f; bi[t] = 0; }
  for (int j = 0; j < NPTS; ++j) {
    float cd = row[j];
    if (cd < bd[KNN - 1]) {
      int ci = j;
      #pragma unroll
      for (int t = 0; t < KNN; ++t) {
        if (cd < bd[t]) {
          float tf = bd[t]; bd[t] = cd; cd = tf;
          int   ti = bi[t]; bi[t] = ci; ci = ti;
        }
      }
    }
  }
  #pragma unroll
  for (int t = 0; t < KNN; ++t) idxb[i * KNN + t] = bi[t];
}

// ---------------- EdgeConv: gather -> [nbr-ctr, ctr] -> 1x1 conv (bf16 WMMA) -> max_k -> leaky ----------------
template<int CIN, int COUT>
__global__ void edgeconv_kernel(const float* __restrict__ X, const int* __restrict__ idx,
                                const __bf16* __restrict__ Wb, const float* __restrict__ bias,
                                float* __restrict__ H) {
  constexpr int C2   = 2 * CIN;
  constexpr int COLS = ((C2 + 31) / 32) * 32;   // K padded to chunk multiple
  __shared__ __bf16 feat[32 * COLS];            // 32 edge rows (20 valid), bf16
  int n = blockIdx.x, b = blockIdx.y;
  const float* Xc = X  + ((size_t)b * NPTS + n) * CIN;
  const int*   ip = idx + ((size_t)b * NPTS + n) * KNN;
  int tid = threadIdx.x;
  for (int i = tid; i < 32 * COLS; i += blockDim.x) {
    int k = i / COLS, c = i % COLS;              // COLS is a power of two
    float v = 0.f;
    if (k < KNN && c < C2) {
      const float* Xn = X + ((size_t)b * NPTS + ip[k]) * CIN;
      v = (c < CIN) ? (Xn[c] - Xc[c]) : Xc[c - CIN];
    }
    feat[i] = (__bf16)v;
  }
  __syncthreads();

  int wave = tid >> 5, lane = tid & 31;
  int m    = lane & 15;
  int khiA = (lane & 16) ? 8 : 0;     // 16-bit A layout: hi lanes offset K by 8
  int khiB = (lane & 16) ? 8 : 0;     // 16-bit B layout: hi lanes offset K by 8 (same as A)
  int o    = wave * 16 + m;           // output channel (B/D column)
  v8f acc0 = {0.f,0.f,0.f,0.f,0.f,0.f,0.f,0.f};
  v8f acc1 = acc0;
  #pragma unroll
  for (int kk = 0; kk < COLS; kk += 32) {
    // A fragment: two contiguous 8-element runs per lane (ds_load_b128 x2 per tile)
    const __bf16* a0p = feat + m * COLS + kk + khiA;
    const __bf16* a1p = feat + (m + 16) * COLS + kk + khiA;
    v16bf a0 = frag16(*(const v8bf*)(a0p), *(const v8bf*)(a0p + 16));
    v16bf a1 = frag16(*(const v8bf*)(a1p), *(const v8bf*)(a1p + 16));
    // B fragment: two contiguous 8-element runs per lane (global_load_b128 x2)
    const __bf16* wp = Wb + (size_t)o * COLS + kk + khiB;
    v16bf bm = frag16(*(const v8bf*)(wp), *(const v8bf*)(wp + 16));
    acc0 = __builtin_amdgcn_wmma_f32_16x16x32_bf16(false, a0, false, bm, (short)0, acc0,
                                                   false, false);
    acc1 = __builtin_amdgcn_wmma_f32_16x16x32_bf16(false, a1, false, bm, (short)0, acc1,
                                                   false, false);
  }
  // max over K edges (leaky is monotone -> activation after max)
  float mx = acc0[0];
  #pragma unroll
  for (int e = 1; e < 8; ++e) mx = fmaxf(mx, acc0[e]);
  mx = fmaxf(mx, __shfl_xor(mx, 16, 32));               // rows 0..15 done
  float m1 = fmaxf(fmaxf(acc1[0], acc1[1]), fmaxf(acc1[2], acc1[3])); // rows 16..19 (lanes 0-15)
  m1 = __shfl(m1, m, 32);                               // broadcast valid half
  float z = fmaxf(mx, m1) + bias[o];
  if (lane < 16) H[((size_t)b * NPTS + n) * COUT + o] = leaky(z);
}

// ---------------- final 512->1024 projection (bf16 WMMA), partial max over 16-row tiles ----------------
__global__ void finalgemm_kernel(const float* __restrict__ h1, const float* __restrict__ h2,
                                 const float* __restrict__ h3, const float* __restrict__ h4,
                                 const __bf16* __restrict__ Wbf, float* __restrict__ zmax) {
  __shared__ __bf16 arow[16 * CCAT];
  int rt = blockIdx.y, b = blockIdx.z;
  int tid = threadIdx.x;
  for (int i = tid; i < 16 * CCAT; i += blockDim.x) {
    int r = i / CCAT, c = i % CCAT;
    size_t p = (size_t)b * NPTS + rt * 16 + r;
    float v;
    if      (c < 64)  v = h1[p * 64  + c];
    else if (c < 128) v = h2[p * 64  + (c - 64)];
    else if (c < 256) v = h3[p * 128 + (c - 128)];
    else              v = h4[p * 256 + (c - 256)];
    arow[i] = (__bf16)v;
  }
  __syncthreads();

  int wave = tid >> 5, lane = tid & 31;
  int m    = lane & 15;
  int khiA = (lane & 16) ? 8 : 0;
  int khiB = (lane & 16) ? 8 : 0;
  int o    = (blockIdx.x * WPB + wave) * 16 + m;
  v8f acc = {0.f,0.f,0.f,0.f,0.f,0.f,0.f,0.f};
  #pragma unroll
  for (int kk = 0; kk < CCAT; kk += 32) {
    const __bf16* ap = arow + m * CCAT + kk + khiA;
    v16bf a = frag16(*(const v8bf*)(ap), *(const v8bf*)(ap + 16));
    const __bf16* wp = Wbf + (size_t)o * CCAT + kk + khiB;
    v16bf bm = frag16(*(const v8bf*)(wp), *(const v8bf*)(wp + 16));
    acc = __builtin_amdgcn_wmma_f32_16x16x32_bf16(false, a, false, bm, (short)0, acc,
                                                  false, false);
  }
  float mx = acc[0];
  #pragma unroll
  for (int e = 1; e < 8; ++e) mx = fmaxf(mx, acc[e]);
  mx = fmaxf(mx, __shfl_xor(mx, 16, 32));   // max over the 16 points of this tile
  if (lane < 16) zmax[((size_t)b * RT + rt) * FDIM + o] = mx;
}

// ---------------- reduce partial maxima over N, add bias, activate ----------------
__global__ void finalreduce_kernel(const float* __restrict__ zmax, const float* __restrict__ bf,
                                   float* __restrict__ out) {
  int o = blockIdx.x * blockDim.x + threadIdx.x;
  int b = blockIdx.y;
  float mx = -3.4e38f;
  for (int rt = 0; rt < RT; ++rt)
    mx = fmaxf(mx, zmax[((size_t)b * RT + rt) * FDIM + o]);
  out[(size_t)b * FDIM + o] = leaky(mx + bf[o]);
}

// ---------------- host-side per-layer kNN phase ----------------
template<int C>
static void knn_phase(const float* in, float* xx, float* d2, int* idx, hipStream_t stream) {
  norms_kernel<C><<<BATCH * NPTS / 256, 256, 0, stream>>>(in, xx);
  dim3 distGrid(NPTS / 16, NPTS / 16);
  for (int b = 0; b < BATCH; ++b) {
    dist_kernel<C><<<distGrid, 32, 0, stream>>>(in + (size_t)b * NPTS * C,
                                                xx + (size_t)b * NPTS, d2);
    topk_kernel<<<NPTS / 256, 256, 0, stream>>>(d2, idx + (size_t)b * NPTS * KNN);
  }
}

extern "C" void kernel_launch(void* const* d_in, const int* in_sizes, int n_in,
                              void* d_out, int out_size, void* d_ws, size_t ws_size,
                              hipStream_t stream) {
  (void)in_sizes; (void)n_in; (void)out_size; (void)ws_size;
  const float* x  = (const float*)d_in[0];
  const float* W1 = (const float*)d_in[1];
  const float* b1 = (const float*)d_in[2];
  const float* W2 = (const float*)d_in[3];
  const float* b2 = (const float*)d_in[4];
  const float* W3 = (const float*)d_in[5];
  const float* b3 = (const float*)d_in[6];
  const float* W4 = (const float*)d_in[7];
  const float* b4 = (const float*)d_in[8];
  const float* Wf = (const float*)d_in[9];
  const float* bf = (const float*)d_in[10];

  char* ws = (char*)d_ws;
  size_t off = 0;
  float*  d2   = (float*)(ws + off); off += (size_t)NPTS * NPTS * 4;         // 16 MB, reused per batch
  float*  xx   = (float*)(ws + off); off += (size_t)BATCH * NPTS * 4;
  int*    idx  = (int*)  (ws + off); off += (size_t)BATCH * NPTS * KNN * 4;
  float*  h1   = (float*)(ws + off); off += (size_t)BATCH * NPTS * 64  * 4;
  float*  h2   = (float*)(ws + off); off += (size_t)BATCH * NPTS * 64  * 4;
  float*  h3   = (float*)(ws + off); off += (size_t)BATCH * NPTS * 128 * 4;
  float*  h4   = (float*)(ws + off); off += (size_t)BATCH * NPTS * 256 * 4;
  float*  zmax = (float*)(ws + off); off += (size_t)BATCH * RT * FDIM * 4;
  __bf16* Wb1  = (__bf16*)(ws + off); off += (size_t)64   * 32  * 2;
  __bf16* Wb2  = (__bf16*)(ws + off); off += (size_t)64   * 128 * 2;
  __bf16* Wb3  = (__bf16*)(ws + off); off += (size_t)128  * 128 * 2;
  __bf16* Wb4  = (__bf16*)(ws + off); off += (size_t)256  * 256 * 2;
  __bf16* Wbf  = (__bf16*)(ws + off); off += (size_t)FDIM * CCAT * 2;

  // Pre-convert weights to zero-padded bf16 (branch-free GEMM inner loops)
  auto convw = [&](const float* W, __bf16* Wb, int O, int C2, int COLS) {
    int total = O * COLS;
    convw_kernel<<<(total + 255) / 256, 256, 0, stream>>>(W, Wb, C2, COLS, total);
  };
  convw(W1, Wb1, 64,   6,   32);
  convw(W2, Wb2, 64,   128, 128);
  convw(W3, Wb3, 128,  128, 128);
  convw(W4, Wb4, 256,  256, 256);
  convw(Wf, Wbf, FDIM, CCAT, CCAT);

  // Layer 1: 3 -> 64
  knn_phase<3>(x, xx, d2, idx, stream);
  edgeconv_kernel<3, 64><<<dim3(NPTS, BATCH), (64 / 16) * 32, 0, stream>>>(x, idx, Wb1, b1, h1);
  // Layer 2: 64 -> 64
  knn_phase<64>(h1, xx, d2, idx, stream);
  edgeconv_kernel<64, 64><<<dim3(NPTS, BATCH), (64 / 16) * 32, 0, stream>>>(h1, idx, Wb2, b2, h2);
  // Layer 3: 64 -> 128
  knn_phase<64>(h2, xx, d2, idx, stream);
  edgeconv_kernel<64, 128><<<dim3(NPTS, BATCH), (128 / 16) * 32, 0, stream>>>(h2, idx, Wb3, b3, h3);
  // Layer 4: 128 -> 256
  knn_phase<128>(h3, xx, d2, idx, stream);
  edgeconv_kernel<128, 256><<<dim3(NPTS, BATCH), (256 / 16) * 32, 0, stream>>>(h3, idx, Wb4, b4, h4);

  // Final projection + global max pool
  finalgemm_kernel<<<dim3(FDIM / (16 * WPB), RT, BATCH), WPB * 32, 0, stream>>>(
      h1, h2, h3, h4, Wbf, zmax);
  finalreduce_kernel<<<dim3(FDIM / 256, BATCH), 256, 0, stream>>>(zmax, bf, (float*)d_out);
}
